// STGATBlock_73521250173075
// MI455X (gfx1250) — matrix operand order repaired and checked by reading, not compiled, not run on hardware
//
#include <hip/hip_runtime.h>
#include <stdint.h>

typedef __attribute__((ext_vector_type(16))) __bf16 v16bf;
typedef __attribute__((ext_vector_type(8)))  float  v8f;

#define S_LENC 12
#define T_LENC 4
#define NNODES 1024
#define E_INC  64
#define E_PREDC 64
#define DMODEL 512
#define NHEADS 8
#define DHEAD  64
#define DFFC   2048
#define NLAY   4
#define GHEADS 8
#define GCH    64
#define GATIN  256
#define GATOUT 256

// ---------------------------------------------------------------- helpers
__device__ __forceinline__ uint16_t f2bf(float f) {
  uint32_t u = __float_as_uint(f);
  u += 0x7FFFu + ((u >> 16) & 1u);
  return (uint16_t)(u >> 16);
}

__device__ __forceinline__ void atomicMaxF(float* addr, float val) {
  int old = __float_as_int(*addr);
  while (__int_as_float(old) < val) {
    int assumed = old;
    old = atomicCAS((int*)addr, assumed, __float_as_int(val));
    if (old == assumed) break;
  }
}

// bf16 fragment held as two 128-bit registers (16 x bf16)
struct Frag { uint4 q0, q1; };
__device__ __forceinline__ void loadFA(Frag& f, const uint16_t* p) {
  f.q0 = *(const uint4*)p;          // elems 0..7  = K kb..kb+7
  f.q1 = *(const uint4*)(p + 16);   // elems 8..15 = K kb+16..kb+23
}
__device__ __forceinline__ void loadFB(Frag& f, const uint16_t* p) {
  f.q0 = *(const uint4*)p;          // elems 0..7  = K kb..kb+7
  f.q1 = *(const uint4*)(p + 16);   // elems 8..15 = K kb+16..kb+23
}
__device__ __forceinline__ v16bf asv(const Frag& f) {
  union { Frag f; v16bf v; } u;
  u.f = f;
  return u.v;
}

// ------------------------------------------------------- elementwise kernels
__global__ void f32_to_bf16_k(const float* __restrict__ in, uint16_t* __restrict__ out, int n) {
  int i = blockIdx.x * 256 + threadIdx.x;
  if (i < n) out[i] = f2bf(in[i]);
}

// W is (K,N) row-major f32; Wt is (N,K) row-major bf16
__global__ void transpose_bf16_k(const float* __restrict__ W, uint16_t* __restrict__ Wt,
                                 int K, int N) {
  int i = blockIdx.x * 256 + threadIdx.x;
  if (i >= K * N) return;
  int n = i / K, k = i - n * K;
  Wt[i] = f2bf(W[(size_t)k * N + n]);
}

__global__ void fill_k(float* p, float v, int n) {
  int i = blockIdx.x * 256 + threadIdx.x;
  if (i < n) p[i] = v;
}

__global__ void bias_rows_k(float* __restrict__ out, const float* __restrict__ bias,
                            int n, int F) {
  int i = blockIdx.x * 256 + threadIdx.x;
  if (i < n) out[i] = bias[i % F];
}

__global__ void add_pe_k(float* __restrict__ x, const float* __restrict__ pe,
                         int total, int bnd, int D) {
  int i = blockIdx.x * 256 + threadIdx.x;
  if (i >= total) return;
  int s = i / bnd;
  int d = i % D;
  x[i] += pe[s * D + d];
}

// trg (1,NN,T,E) -> tgt (T*NN, E) with row = t*NN+n
__global__ void reshape_tgt_k(const float* __restrict__ trg, float* __restrict__ tgt) {
  int i = blockIdx.x * 256 + threadIdx.x;
  if (i >= T_LENC * NNODES * E_PREDC) return;
  int e = i % E_PREDC;
  int n = (i / E_PREDC) % NNODES;
  int t = i / (E_PREDC * NNODES);
  tgt[i] = trg[((size_t)n * T_LENC + t) * E_PREDC + e];
}

// tout (T*NN, E) -> tg (NN, T*E)
__global__ void reshape_tout_k(const float* __restrict__ tin, float* __restrict__ tg) {
  int i = blockIdx.x * 256 + threadIdx.x;
  if (i >= NNODES * GATIN) return;
  int j = i % GATIN;
  int n = i / GATIN;
  int t = j / E_PREDC, e = j % E_PREDC;
  tg[i] = tin[((size_t)t * NNODES + n) * E_PREDC + e];
}

// ------------------------------------------------------------- layernorm
// out = LN(x (+ r)) * g + b ; in-place safe (stages row in LDS). D <= 512.
__global__ void add_ln_k(const float* __restrict__ x, const float* __restrict__ r,
                         const float* __restrict__ g, const float* __restrict__ b,
                         float* __restrict__ out, int D) {
  __shared__ float sv[512];
  __shared__ float red[256];
  int row = blockIdx.x, tid = threadIdx.x;
  const float* xp = x + (size_t)row * D;
  const float* rp = r ? r + (size_t)row * D : nullptr;
  float ls = 0.f;
  for (int i = tid; i < D; i += 256) {
    float v = xp[i] + (rp ? rp[i] : 0.f);
    sv[i] = v;
    ls += v;
  }
  red[tid] = ls;
  __syncthreads();
  for (int s = 128; s > 0; s >>= 1) { if (tid < s) red[tid] += red[tid + s]; __syncthreads(); }
  float mean = red[0] / D;
  __syncthreads();
  float lv = 0.f;
  for (int i = tid; i < D; i += 256) { float d = sv[i] - mean; lv += d * d; }
  red[tid] = lv;
  __syncthreads();
  for (int s = 128; s > 0; s >>= 1) { if (tid < s) red[tid] += red[tid + s]; __syncthreads(); }
  float inv = rsqrtf(red[0] / D + 1e-5f);
  for (int i = tid; i < D; i += 256)
    out[(size_t)row * D + i] = (sv[i] - mean) * inv * g[i] + b[i];
}

// ---------------------------------------------------------------- attention
// Q rows = q*NN+b, K/V rows = k*NN+b, stride DMODEL; one block per (b,h), 64 thr.
__global__ __launch_bounds__(64)
void attn_k(const float* __restrict__ Q, const float* __restrict__ Km,
            const float* __restrict__ Vm, const float* __restrict__ mask,
            float* __restrict__ O, int Lq, int Lk) {
  const int b = blockIdx.x >> 3;
  const int h = blockIdx.x & 7;
  const int tid = threadIdx.x;
  __shared__ float sQ[S_LENC * DHEAD], sK[S_LENC * DHEAD], sV[S_LENC * DHEAD];
  __shared__ float sS[S_LENC * S_LENC];
  for (int q = 0; q < Lq; ++q)
    sQ[q * DHEAD + tid] = Q[((size_t)(q * NNODES + b)) * DMODEL + h * DHEAD + tid];
  for (int k = 0; k < Lk; ++k) {
    sK[k * DHEAD + tid] = Km[((size_t)(k * NNODES + b)) * DMODEL + h * DHEAD + tid];
    sV[k * DHEAD + tid] = Vm[((size_t)(k * NNODES + b)) * DMODEL + h * DHEAD + tid];
  }
  __syncthreads();
  for (int idx = tid; idx < Lq * Lk; idx += 64) {
    int q = idx / Lk, k = idx % Lk;
    float s = 0.f;
    for (int d = 0; d < DHEAD; ++d) s += sQ[q * DHEAD + d] * sK[k * DHEAD + d];
    s *= 0.125f;
    if (mask) s += mask[q * Lk + k];
    sS[idx] = s;
  }
  __syncthreads();
  if (tid < Lq) {
    float mx = -3e38f;
    for (int k = 0; k < Lk; ++k) mx = fmaxf(mx, sS[tid * Lk + k]);
    float sum = 0.f;
    for (int k = 0; k < Lk; ++k) { float e = __expf(sS[tid * Lk + k] - mx); sS[tid * Lk + k] = e; sum += e; }
    float inv = 1.f / sum;
    for (int k = 0; k < Lk; ++k) sS[tid * Lk + k] *= inv;
  }
  __syncthreads();
  for (int q = 0; q < Lq; ++q) {
    float acc = 0.f;
    for (int k = 0; k < Lk; ++k) acc += sS[q * Lk + k] * sV[k * DHEAD + tid];
    O[((size_t)(q * NNODES + b)) * DMODEL + h * DHEAD + tid] = acc;
  }
}

// ------------------------------------------------------------- WMMA GEMM
// C[M,N] = act(A[M,K](bf16) * Bt[N,K]^T(bf16) + bias); block=128=4 waves,
// each wave a 16x64 strip (4 accumulators). K % 64 == 0, M % 32 == 0.
// Wave id forced into SGPR via readfirstlane -> all guards are scalar
// branches, EXEC stays all-ones around every WMMA. The K-loop is
// double-buffered (ping-pong fragment sets) so fragment loads for the next
// 32-wide K slice are in flight while WMMAs consume the current one,
// letting the compiler issue partial s_wait_loadcnt instead of full drains.
template <int ACT, int OBF>
__global__ __launch_bounds__(128)
void wmma_gemm_k(const uint16_t* __restrict__ A, const uint16_t* __restrict__ Bt,
                 const float* __restrict__ bias, float* __restrict__ Cf,
                 uint16_t* __restrict__ Cb, int M, int N, int K) {
  const int lane = threadIdx.x & 31;
  const int w = __builtin_amdgcn_readfirstlane(threadIdx.x) >> 5;  // SGPR wave id
  const int m0 = blockIdx.x * 32 + (w >> 1) * 16;
  const int nb = blockIdx.y * 128 + (w & 1) * 64;
  if (nb >= N) return;                 // scalar: whole wave exits (no barriers here)

  v8f acc[4];
#pragma unroll
  for (int t = 0; t < 4; ++t)
#pragma unroll
    for (int i = 0; i < 8; ++i) acc[t][i] = 0.f;

  const int arow = m0 + (lane & 15);
  const int akb = (lane >> 4) << 3;   // A: elems 0..7 = K kb..kb+7, 8..15 = kb+16..kb+23
  const int bro = lane & 15;
  const int bkb = (lane >> 4) << 3;   // B: elems 0..7 = K kb..kb+7, 8..15 = kb+16..kb+23

  const uint16_t* pa0 = A + (size_t)arow * K + akb;
  const uint16_t* pb0 = Bt + (size_t)(nb + bro) * K + bkb;
  const bool full = (nb + 64 <= N);   // scalar

  if (full) {
    // ---- software-pipelined path (K % 64 == 0) ----
    Frag a0, a1, b0[4], b1[4];
    loadFA(a0, pa0);
#pragma unroll
    for (int t = 0; t < 4; ++t) loadFB(b0[t], pb0 + (size_t)t * 16 * K);
    for (int k0 = 0; k0 < K; k0 += 64) {
      // issue loads for second half of this 64-slice
      loadFA(a1, pa0 + k0 + 32);
#pragma unroll
      for (int t = 0; t < 4; ++t) loadFB(b1[t], pb0 + (size_t)t * 16 * K + k0 + 32);
      // compute on buffer 0 while buffer 1 loads are in flight
#pragma unroll
      for (int t = 0; t < 4; ++t)
        acc[t] = __builtin_amdgcn_wmma_f32_16x16x32_bf16(
            false, asv(a0), false, asv(b0[t]), (short)0, acc[t], false, false);
      // issue loads for next 64-slice (scalar condition)
      if (k0 + 64 < K) {
        loadFA(a0, pa0 + k0 + 64);
#pragma unroll
        for (int t = 0; t < 4; ++t) loadFB(b0[t], pb0 + (size_t)t * 16 * K + k0 + 64);
      }
      // compute on buffer 1 while buffer 0 loads are in flight
#pragma unroll
      for (int t = 0; t < 4; ++t)
        acc[t] = __builtin_amdgcn_wmma_f32_16x16x32_bf16(
            false, asv(a1), false, asv(b1[t]), (short)0, acc[t], false, false);
    }
  } else {
    // ---- guarded fallback (partial N tile; scalar guards) ----
    for (int k0 = 0; k0 < K; k0 += 32) {
      Frag fa;
      loadFA(fa, pa0 + k0);
#pragma unroll
      for (int t = 0; t < 4; ++t) {
        if (nb + t * 16 < N) {
          Frag fb;
          loadFB(fb, pb0 + (size_t)t * 16 * K + k0);
          acc[t] = __builtin_amdgcn_wmma_f32_16x16x32_bf16(
              false, asv(fa), false, asv(fb), (short)0, acc[t], false, false);
        }
      }
    }
  }

  const int rbase = m0 + ((lane >> 4) << 3);
#pragma unroll
  for (int t = 0; t < 4; ++t) {
    int n0 = nb + t * 16;
    if (n0 >= N) continue;            // scalar
    int col = n0 + (lane & 15);
    float bv = bias ? bias[col] : 0.f;
#pragma unroll
    for (int vv = 0; vv < 8; ++vv) {
      float xv = acc[t][vv] + bv;
      if (ACT == 1) xv = fmaxf(xv, 0.f);
      if (ACT == 2) xv = 1.f / (1.f + __expf(-xv));
      size_t o = (size_t)(rbase + vv) * N + col;
      if (OBF) Cb[o] = f2bf(xv);
      else Cf[o] = xv;
    }
  }
}

// ------------------------------------------------------------- GATv2 kernels
__global__ void gat_alpha_k(const float* __restrict__ xl, const float* __restrict__ xr,
                            const int* __restrict__ ei, const float* __restrict__ att,
                            float* __restrict__ alpha, float* __restrict__ amax,
                            int E0, int E2, int H, int C) {
  int idx = blockIdx.x * 256 + threadIdx.x;
  if (idx >= E2 * H) return;
  int e = idx / H, h = idx - e * H;
  int s = e < E0 ? ei[e] : (e - E0);
  int d = e < E0 ? ei[E0 + e] : (e - E0);
  int HC = H * C;
  const float* pl = xl + (size_t)s * HC + h * C;
  const float* pr = xr + (size_t)d * HC + h * C;
  const float* pa = att + h * C;
  float acc = 0.f;
  for (int c = 0; c < C; ++c) {
    float m = pl[c] + pr[c];
    m = m > 0.f ? m : 0.2f * m;   // leaky_relu(0.2)
    acc += m * pa[c];
  }
  alpha[idx] = acc;
  atomicMaxF(&amax[d * H + h], acc);
}

__global__ void gat_ex_k(const float* __restrict__ alpha, const int* __restrict__ ei,
                         const float* __restrict__ amax, float* __restrict__ ex,
                         float* __restrict__ denom, int E0, int E2, int H) {
  int idx = blockIdx.x * 256 + threadIdx.x;
  if (idx >= E2 * H) return;
  int e = idx / H, h = idx - e * H;
  int d = e < E0 ? ei[E0 + e] : (e - E0);
  float v = __expf(alpha[idx] - amax[d * H + h]);
  ex[idx] = v;
  atomicAdd(&denom[d * H + h], v);
}

__global__ void gat_scatter_k(const float* __restrict__ ex, const float* __restrict__ denom,
                              const float* __restrict__ xl, const int* __restrict__ ei,
                              float* __restrict__ out, int E0, int E2, int H, int C) {
  int idx = blockIdx.x * 256 + threadIdx.x;
  int HC = H * C;
  if (idx >= E2 * HC) return;
  int e = idx / HC, j = idx - e * HC;
  int h = j / C;
  int s = e < E0 ? ei[e] : (e - E0);
  int d = e < E0 ? ei[E0 + e] : (e - E0);
  float wv = ex[e * H + h] / (denom[d * H + h] + 1e-16f);
  atomicAdd(&out[(size_t)d * HC + j], wv * xl[(size_t)s * HC + j]);
}

__global__ void rownorm_k(const float* __restrict__ h, float* __restrict__ z,
                          uint16_t* __restrict__ zb, int D) {
  __shared__ float red[256];
  int row = blockIdx.x, tid = threadIdx.x;   // D == 256 == blockDim
  float v = h[(size_t)row * D + tid];
  red[tid] = v * v;
  __syncthreads();
  for (int s = 128; s > 0; s >>= 1) { if (tid < s) red[tid] += red[tid + s]; __syncthreads(); }
  float nrm = fmaxf(sqrtf(red[0]), 1e-12f);
  float zz = v / nrm;
  z[(size_t)row * D + tid] = zz;
  zb[(size_t)row * D + tid] = f2bf(zz);
}

// ================================================================== host
extern "C" void kernel_launch(void* const* d_in, const int* in_sizes, int n_in,
                              void* d_out, int out_size, void* d_ws, size_t ws_size,
                              hipStream_t stream) {
  (void)n_in; (void)out_size; (void)ws_size;
  const float* src     = (const float*)d_in[0];
  const float* trg     = (const float*)d_in[1];
  const float* srcmask = (const float*)d_in[2];
  const float* tgtmask = (const float*)d_in[3];
  const int*   eidx    = (const int*)d_in[4];
  const int E0 = in_sizes[4] / 2;
  const int E2 = E0 + NNODES;

  auto FP = [&](int i) { return (const float*)d_in[i]; };
  // jax pytree (sorted-key) leaf order
  const int DEC_IN_W = 5, DEC_IN_B = 6;
  auto DL = [&](int l, int o) { return 7 + l * 26 + o; };
  // dec layer: ca{Wk,Wo,Wq,Wv,bk,bo,bq,bv}=0..7 ffn{W1,W2,b1,b2}=8..11
  // ln1{b,g}=12,13 ln2=14,15 ln3=16,17 sa{Wk..bv}=18..25
  const int DEC_NORM_B = 111, DEC_NORM_G = 112;
  const int ENC_IN_W = 113, ENC_IN_B = 114;
  auto EL = [&](int l, int o) { return 115 + l * 16 + o; };
  // enc layer: attn{Wk,Wo,Wq,Wv,bk,bo,bq,bv}=0..7 ffn=8..11 ln1=12,13 ln2=14,15
  const int ENC_NORM_B = 179, ENC_NORM_G = 180;
  const int G1_WL = 181, G1_WR = 182, G1_ATT = 183, G1_B = 184;
  const int G2_WL = 185, G2_WR = 186, G2_ATT = 187, G2_B = 188;
  const int LIN_W = 189, LIN_B = 190, PE = 191;

  // ---- workspace carve
  char* wsb = (char*)d_ws;
  size_t off = 0;
  auto alloc = [&](size_t bytes) -> void* {
    off = (off + 255) & ~(size_t)255;
    void* p = wsb + off;
    off += bytes;
    return p;
  };
  const int MT = S_LENC * NNODES;   // 12288 encoder tokens
  const int MD = T_LENC * NNODES;   // 4096 decoder tokens
  float* x  = (float*)alloc((size_t)MT * DMODEL * 4);   // enc state / mem
  float* qb = (float*)alloc((size_t)MT * DMODEL * 4);
  float* kb = (float*)alloc((size_t)MT * DMODEL * 4);
  float* vb = (float*)alloc((size_t)MT * DMODEL * 4);
  float* ao = (float*)alloc((size_t)MT * DMODEL * 4);
  float* y  = (float*)alloc((size_t)MD * DMODEL * 4);
  uint16_t* xb   = (uint16_t*)alloc((size_t)MT * DMODEL * 2);
  uint16_t* hb   = (uint16_t*)alloc((size_t)MT * DFFC * 2);
  uint16_t* memb = (uint16_t*)alloc((size_t)MT * DMODEL * 2);
  uint16_t* wts  = (uint16_t*)alloc((size_t)DMODEL * DFFC * 2); // weight slot
  float* tout = (float*)alloc((size_t)MD * E_PREDC * 4);
  float* tg   = (float*)alloc((size_t)NNODES * GATIN * 4);
  uint16_t* tgb = (uint16_t*)alloc((size_t)NNODES * GATIN * 2);
  float* xl1 = (float*)alloc((size_t)NNODES * 512 * 4);
  float* xr1 = (float*)alloc((size_t)NNODES * 512 * 4);
  float* h1  = (float*)alloc((size_t)NNODES * 512 * 4);
  uint16_t* h1b = (uint16_t*)alloc((size_t)NNODES * 512 * 2);
  float* al1 = (float*)alloc((size_t)E2 * GHEADS * 4);
  float* ex1 = (float*)alloc((size_t)E2 * GHEADS * 4);
  float* am1 = (float*)alloc((size_t)NNODES * GHEADS * 4);
  float* dn1 = (float*)alloc((size_t)NNODES * GHEADS * 4);
  float* xl2 = (float*)alloc((size_t)NNODES * GATOUT * 4);
  float* xr2 = (float*)alloc((size_t)NNODES * GATOUT * 4);
  float* h2  = (float*)alloc((size_t)NNODES * GATOUT * 4);
  float* al2 = (float*)alloc((size_t)E2 * 4);
  float* ex2 = (float*)alloc((size_t)E2 * 4);
  float* am2 = (float*)alloc((size_t)NNODES * 4);
  float* dn2 = (float*)alloc((size_t)NNODES * 4);
  uint16_t* zb = (uint16_t*)alloc((size_t)NNODES * GATOUT * 2);

  // ---- launch helpers (stream-ordered; wts slot reused serially)
  auto toBf = [&](const float* in, uint16_t* o, int n) {
    f32_to_bf16_k<<<dim3((n + 255) / 256), dim3(256), 0, stream>>>(in, o, n);
  };
  auto convW = [&](const float* W, int K, int N) {
    int n = K * N;
    transpose_bf16_k<<<dim3((n + 255) / 256), dim3(256), 0, stream>>>(W, wts, K, N);
  };
  auto gemmF = [&](const uint16_t* A, const uint16_t* Bt, const float* bias, float* C,
                   int M, int N, int K) {
    wmma_gemm_k<0, 0><<<dim3(M / 32, (N + 127) / 128), dim3(128), 0, stream>>>(
        A, Bt, bias, C, (uint16_t*)nullptr, M, N, K);
  };
  auto gemmReluB = [&](const uint16_t* A, const uint16_t* Bt, const float* bias, uint16_t* C,
                       int M, int N, int K) {
    wmma_gemm_k<1, 1><<<dim3(M / 32, (N + 127) / 128), dim3(128), 0, stream>>>(
        A, Bt, bias, (float*)nullptr, C, M, N, K);
  };
  auto addLN = [&](const float* xin, const float* r, int gi, int bi, float* o, int M) {
    add_ln_k<<<dim3(M), dim3(256), 0, stream>>>(xin, r, FP(gi), FP(bi), o, DMODEL);
  };
  auto attn = [&](const float* Qp, const float* Kp, const float* Vp, const float* m,
                  float* Op, int Lq, int Lk) {
    attn_k<<<dim3(NNODES * NHEADS), dim3(64), 0, stream>>>(Qp, Kp, Vp, m, Op, Lq, Lk);
  };

  // ============================ encoder ============================
  toBf(src, xb, MT * E_INC);
  convW(FP(ENC_IN_W), E_INC, DMODEL);
  gemmF(xb, wts, FP(ENC_IN_B), x, MT, DMODEL, E_INC);
  add_pe_k<<<dim3((MT * DMODEL + 255) / 256), dim3(256), 0, stream>>>(
      x, FP(PE), MT * DMODEL, NNODES * DMODEL, DMODEL);

  for (int l = 0; l < NLAY; ++l) {
    toBf(x, xb, MT * DMODEL);
    convW(FP(EL(l, 2)), DMODEL, DMODEL); gemmF(xb, wts, FP(EL(l, 6)), qb, MT, DMODEL, DMODEL); // Wq
    convW(FP(EL(l, 0)), DMODEL, DMODEL); gemmF(xb, wts, FP(EL(l, 4)), kb, MT, DMODEL, DMODEL); // Wk
    convW(FP(EL(l, 3)), DMODEL, DMODEL); gemmF(xb, wts, FP(EL(l, 7)), vb, MT, DMODEL, DMODEL); // Wv
    attn(qb, kb, vb, nullptr, ao, S_LENC, S_LENC);
    toBf(ao, xb, MT * DMODEL);
    convW(FP(EL(l, 1)), DMODEL, DMODEL); gemmF(xb, wts, FP(EL(l, 5)), qb, MT, DMODEL, DMODEL); // Wo
    addLN(x, qb, EL(l, 13), EL(l, 12), x, MT);                                                 // ln1
    toBf(x, xb, MT * DMODEL);
    convW(FP(EL(l, 8)), DMODEL, DFFC); gemmReluB(xb, wts, FP(EL(l, 10)), hb, MT, DFFC, DMODEL); // W1+relu
    convW(FP(EL(l, 9)), DFFC, DMODEL); gemmF(hb, wts, FP(EL(l, 11)), qb, MT, DMODEL, DFFC);     // W2
    addLN(x, qb, EL(l, 15), EL(l, 14), x, MT);                                                  // ln2
  }
  addLN(x, nullptr, ENC_NORM_G, ENC_NORM_B, x, MT);  // mem = enc_norm(x), in-place
  toBf(x, memb, MT * DMODEL);

  // ============================ decoder ============================
  reshape_tgt_k<<<dim3((MD * E_PREDC + 255) / 256), dim3(256), 0, stream>>>(trg, tout);
  toBf(tout, xb, MD * E_PREDC);
  convW(FP(DEC_IN_W), E_PREDC, DMODEL);
  gemmF(xb, wts, FP(DEC_IN_B), y, MD, DMODEL, E_PREDC);
  add_pe_k<<<dim3((MD * DMODEL + 255) / 256), dim3(256), 0, stream>>>(
      y, FP(PE), MD * DMODEL, NNODES * DMODEL, DMODEL);

  for (int l = 0; l < NLAY; ++l) {
    // self-attention (causal)
    toBf(y, xb, MD * DMODEL);
    convW(FP(DL(l, 20)), DMODEL, DMODEL); gemmF(xb, wts, FP(DL(l, 24)), qb, MD, DMODEL, DMODEL);
    convW(FP(DL(l, 18)), DMODEL, DMODEL); gemmF(xb, wts, FP(DL(l, 22)), kb, MD, DMODEL, DMODEL);
    convW(FP(DL(l, 21)), DMODEL, DMODEL); gemmF(xb, wts, FP(DL(l, 25)), vb, MD, DMODEL, DMODEL);
    attn(qb, kb, vb, tgtmask, ao, T_LENC, T_LENC);
    toBf(ao, xb, MD * DMODEL);
    convW(FP(DL(l, 19)), DMODEL, DMODEL); gemmF(xb, wts, FP(DL(l, 23)), qb, MD, DMODEL, DMODEL);
    addLN(y, qb, DL(l, 13), DL(l, 12), y, MD);   // ln1
    // cross-attention vs mem
    toBf(y, xb, MD * DMODEL);
    convW(FP(DL(l, 2)), DMODEL, DMODEL); gemmF(xb, wts, FP(DL(l, 6)), qb, MD, DMODEL, DMODEL);
    convW(FP(DL(l, 0)), DMODEL, DMODEL); gemmF(memb, wts, FP(DL(l, 4)), kb, MT, DMODEL, DMODEL);
    convW(FP(DL(l, 3)), DMODEL, DMODEL); gemmF(memb, wts, FP(DL(l, 7)), vb, MT, DMODEL, DMODEL);
    attn(qb, kb, vb, srcmask, ao, T_LENC, S_LENC);
    toBf(ao, xb, MD * DMODEL);
    convW(FP(DL(l, 1)), DMODEL, DMODEL); gemmF(xb, wts, FP(DL(l, 5)), qb, MD, DMODEL, DMODEL);
    addLN(y, qb, DL(l, 15), DL(l, 14), y, MD);   // ln2
    // FFN
    toBf(y, xb, MD * DMODEL);
    convW(FP(DL(l, 8)), DMODEL, DFFC); gemmReluB(xb, wts, FP(DL(l, 10)), hb, MD, DFFC, DMODEL);
    convW(FP(DL(l, 9)), DFFC, DMODEL); gemmF(hb, wts, FP(DL(l, 11)), qb, MD, DMODEL, DFFC);
    addLN(y, qb, DL(l, 17), DL(l, 16), y, MD);   // ln3
  }
  addLN(y, nullptr, DEC_NORM_G, DEC_NORM_B, y, MD);
  toBf(y, xb, MD * DMODEL);
  convW(FP(LIN_W), DMODEL, E_PREDC);
  gemmF(xb, wts, FP(LIN_B), tout, MD, E_PREDC, DMODEL);
  reshape_tout_k<<<dim3((NNODES * GATIN + 255) / 256), dim3(256), 0, stream>>>(tout, tg);
  toBf(tg, tgb, NNODES * GATIN);

  // ============================ GAT layer 1 ============================
  convW(FP(G1_WL), GATIN, GHEADS * GCH); gemmF(tgb, wts, nullptr, xl1, NNODES, GHEADS * GCH, GATIN);
  convW(FP(G1_WR), GATIN, GHEADS * GCH); gemmF(tgb, wts, nullptr, xr1, NNODES, GHEADS * GCH, GATIN);
  fill_k<<<dim3((NNODES * GHEADS + 255) / 256), dim3(256), 0, stream>>>(am1, -3e38f, NNODES * GHEADS);
  fill_k<<<dim3((NNODES * GHEADS + 255) / 256), dim3(256), 0, stream>>>(dn1, 0.f, NNODES * GHEADS);
  bias_rows_k<<<dim3((NNODES * 512 + 255) / 256), dim3(256), 0, stream>>>(h1, FP(G1_B), NNODES * 512, 512);
  gat_alpha_k<<<dim3((E2 * GHEADS + 255) / 256), dim3(256), 0, stream>>>(
      xl1, xr1, eidx, FP(G1_ATT), al1, am1, E0, E2, GHEADS, GCH);
  gat_ex_k<<<dim3((E2 * GHEADS + 255) / 256), dim3(256), 0, stream>>>(
      al1, eidx, am1, ex1, dn1, E0, E2, GHEADS);
  gat_scatter_k<<<dim3((E2 * GHEADS * GCH + 255) / 256), dim3(256), 0, stream>>>(
      ex1, dn1, xl1, eidx, h1, E0, E2, GHEADS, GCH);
  toBf(h1, h1b, NNODES * 512);

  // ============================ GAT layer 2 ============================
  convW(FP(G2_WL), 512, GATOUT); gemmF(h1b, wts, nullptr, xl2, NNODES, GATOUT, 512);
  convW(FP(G2_WR), 512, GATOUT); gemmF(h1b, wts, nullptr, xr2, NNODES, GATOUT, 512);
  fill_k<<<dim3((NNODES + 255) / 256), dim3(256), 0, stream>>>(am2, -3e38f, NNODES);
  fill_k<<<dim3((NNODES + 255) / 256), dim3(256), 0, stream>>>(dn2, 0.f, NNODES);
  bias_rows_k<<<dim3((NNODES * GATOUT + 255) / 256), dim3(256), 0, stream>>>(
      h2, FP(G2_B), NNODES * GATOUT, GATOUT);
  gat_alpha_k<<<dim3((E2 + 255) / 256), dim3(256), 0, stream>>>(
      xl2, xr2, eidx, FP(G2_ATT), al2, am2, E0, E2, 1, GATOUT);
  gat_ex_k<<<dim3((E2 + 255) / 256), dim3(256), 0, stream>>>(
      al2, eidx, am2, ex2, dn2, E0, E2, 1);
  gat_scatter_k<<<dim3((E2 * GATOUT + 255) / 256), dim3(256), 0, stream>>>(
      ex2, dn2, xl2, eidx, h2, E0, E2, 1, GATOUT);

  // ============================ outputs ============================
  float* outp = (float*)d_out;
  float* zf = outp + (size_t)NNODES * NNODES;   // second output: z
  rownorm_k<<<dim3(NNODES), dim3(256), 0, stream>>>(h2, zf, zb, GATOUT);
  // A_pred = sigmoid(z @ z^T): A=z(bf16), Bt=z(bf16)
  wmma_gemm_k<2, 0><<<dim3(NNODES / 32, (NNODES + 127) / 128), dim3(128), 0, stream>>>(
      zb, zb, nullptr, outp, (uint16_t*)nullptr, NNODES, NNODES, GATOUT);
}
